// SparseMoEBlock_515396076110
// MI455X (gfx1250) — matrix operand order repaired and checked by reading, not compiled, not run on hardware
//
#include <hip/hip_runtime.h>
#include <hip/hip_bf16.h>
#include <stdint.h>

#define B_   4
#define T_   1024
#define D_   1024
#define H_   16
#define HS_  64
#define E_   8
#define DFF_ 4096
#define BT_  (B_ * T_)

typedef __bf16 bf16;
typedef __attribute__((ext_vector_type(16))) __bf16 v16bf;
typedef __attribute__((ext_vector_type(8)))  float  v8f;

union FragU { uint4 q[2]; v16bf v; };

__device__ inline v8f wmma_bf16(v16bf a, v16bf b, v8f c) {
  return __builtin_amdgcn_wmma_f32_16x16x32_bf16(false, a, false, b, (short)0, c,
                                                 false, false);
}

// Low 32 bits of a generic LDS pointer are the LDS byte address (ISA 10.2).
__device__ inline uint32_t lds_addr32(const void* p) {
  return (uint32_t)(uintptr_t)p;
}

// CDNA5 async global->LDS copies (ASYNCcnt).  INST_OFFSET is added to BOTH the
// LDS and the global address (ISA 10.7/15.18), so mirrored tiles need one base.
__device__ inline void async_b128(uint32_t lds, const void* g) {
  asm volatile("global_load_async_to_lds_b128 %0, %1, off"
               :: "v"(lds), "v"((uint64_t)(uintptr_t)g) : "memory");
}
__device__ inline void async_b128_x2(uint32_t lds, const void* g) {
  asm volatile("global_load_async_to_lds_b128 %0, %1, off\n\t"
               "global_load_async_to_lds_b128 %0, %1, off offset:16"
               :: "v"(lds), "v"((uint64_t)(uintptr_t)g) : "memory");
}
__device__ inline void async_b128_x4(uint32_t lds, const void* g) {
  asm volatile("global_load_async_to_lds_b128 %0, %1, off\n\t"
               "global_load_async_to_lds_b128 %0, %1, off offset:16\n\t"
               "global_load_async_to_lds_b128 %0, %1, off offset:32\n\t"
               "global_load_async_to_lds_b128 %0, %1, off offset:48"
               :: "v"(lds), "v"((uint64_t)(uintptr_t)g) : "memory");
}
template <int N>
__device__ inline void wait_async() {
  asm volatile("s_wait_asynccnt %0" :: "i"(N) : "memory");
}

// A-matrix 16x32 bf16 fragment: lanes 0-15 hold row M=lane, K0-7 & K16-23;
// lanes 16-31 hold row M=lane-16, K8-15 & K24-31.  LDS tile layout [M][K=stride].
__device__ inline v16bf frag_a(const bf16* base, int row0, int stride, int lane) {
  const int l16 = lane & 15, hf = lane >> 4;
  const bf16* p = base + (size_t)(row0 + l16) * stride + hf * 8;
  FragU f;
  f.q[0] = *(const uint4*)(p);
  f.q[1] = *(const uint4*)(p + 16);
  return f.v;
}

// B-matrix 32x16 bf16 fragment from a [N][K=stride] (B-transposed) tile:
// lanes 0-15 hold col N=lane, K0-7 & K16-23; lanes 16-31 hold K8-15 & K24-31.
__device__ inline v16bf frag_b(const bf16* base, int row0, int stride, int lane) {
  const int l16 = lane & 15, hf = lane >> 4;
  const bf16* p = base + (size_t)(row0 + l16) * stride + hf * 8;
  FragU f;
  f.q[0] = *(const uint4*)(p);
  f.q[1] = *(const uint4*)(p + 16);
  return f.v;
}

// ---------------------------------------------------------------------------
// Generic bf16 WMMA GEMM: C[M,N] = A[M,K] * Bt[N,K]^T  (Bt stored [N][K])
// Block tile 128x128x32, 8 waves, each wave 32x64 (2x4 WMMA tiles).
// Double-buffered LDS fed by async global->LDS copies.
// ---------------------------------------------------------------------------
struct GemmP {
  const bf16* A;
  const bf16* Bt;
  int M, N, K;
  const float* bias;
  const float* resid;
  const float* gate;
  float* outF;
  bf16*  outB;
  bf16*  q;
  bf16*  k;
  bf16*  vT;
  int expert;
};

template <int MODE>
__global__ void __launch_bounds__(256) k_gemm(GemmP p) {
  __shared__ __align__(16) bf16 As[2][128 * 32];
  __shared__ __align__(16) bf16 Bs[2][128 * 32];
  const int bm = blockIdx.y * 128, bn = blockIdx.x * 128;
  const int tid = threadIdx.x, wid = tid >> 5, lane = tid & 31;
  const int wm = (wid & 3) * 32, wn = (wid >> 2) * 64;

  v8f acc[2][4];
#pragma unroll
  for (int i = 0; i < 2; ++i)
#pragma unroll
    for (int j = 0; j < 4; ++j)
#pragma unroll
      for (int r = 0; r < 8; ++r) acc[i][j][r] = 0.f;

  const int ldRow = tid >> 1, ldC = (tid & 1) * 16;
  const bf16* gA = p.A  + (size_t)(bm + ldRow) * p.K + ldC;
  const bf16* gB = p.Bt + (size_t)(bn + ldRow) * p.K + ldC;
  uint32_t ldsA[2], ldsB[2];
  ldsA[0] = lds_addr32(&As[0][ldRow * 32 + ldC]);
  ldsA[1] = lds_addr32(&As[1][ldRow * 32 + ldC]);
  ldsB[0] = lds_addr32(&Bs[0][ldRow * 32 + ldC]);
  ldsB[1] = lds_addr32(&Bs[1][ldRow * 32 + ldC]);

  // prologue: stage tile 0 into buffer 0
  async_b128_x2(ldsA[0], gA);
  async_b128_x2(ldsB[0], gB);

  int buf = 0;
  for (int kk = 0; kk < p.K; kk += 32) {
    if (kk + 32 < p.K) {               // stream next tile into other buffer
      async_b128_x2(ldsA[buf ^ 1], gA + kk + 32);
      async_b128_x2(ldsB[buf ^ 1], gB + kk + 32);
      wait_async<4>();                 // current tile's 4 copies complete
    } else {
      wait_async<0>();
    }
    __syncthreads();                   // all waves see current tile

    const bf16* Ab = As[buf];
    const bf16* Bb = Bs[buf];
    v16bf a0 = frag_a(Ab, wm,      32, lane);
    v16bf a1 = frag_a(Ab, wm + 16, 32, lane);
    v16bf b0 = frag_b(Bb, wn,      32, lane);
    v16bf b1 = frag_b(Bb, wn + 16, 32, lane);
    v16bf b2 = frag_b(Bb, wn + 32, 32, lane);
    v16bf b3 = frag_b(Bb, wn + 48, 32, lane);

    acc[0][0] = wmma_bf16(a0, b0, acc[0][0]);
    acc[0][1] = wmma_bf16(a0, b1, acc[0][1]);
    acc[0][2] = wmma_bf16(a0, b2, acc[0][2]);
    acc[0][3] = wmma_bf16(a0, b3, acc[0][3]);
    acc[1][0] = wmma_bf16(a1, b0, acc[1][0]);
    acc[1][1] = wmma_bf16(a1, b1, acc[1][1]);
    acc[1][2] = wmma_bf16(a1, b2, acc[1][2]);
    acc[1][3] = wmma_bf16(a1, b3, acc[1][3]);

    __syncthreads();                   // all waves done reading this buffer
    buf ^= 1;
  }

  // Epilogue. C layout: lane 0-15 col=lane, VGPR r -> row r; lanes 16-31 row r+8.
  const int l16 = lane & 15, hf = lane >> 4;
#pragma unroll
  for (int i = 0; i < 2; ++i) {
#pragma unroll
    for (int j = 0; j < 4; ++j) {
#pragma unroll
      for (int r = 0; r < 8; ++r) {
        const int gm = bm + wm + i * 16 + r + hf * 8;
        const int gn = bn + wn + j * 16 + l16;
        const float v = acc[i][j][r];
        if constexpr (MODE == 0) {              // QKV scatter
          const int sec = gn >> 10, ns = gn & 1023, h = ns >> 6, e = ns & 63;
          const int b = gm >> 10, t = gm & 1023;
          const bf16 bv = (bf16)v;
          if (sec == 0)
            p.q[(((size_t)(b * H_ + h)) * T_ + t) * HS_ + e] = bv;
          else if (sec == 1)
            p.k[(((size_t)(b * H_ + h)) * T_ + t) * HS_ + e] = bv;
          else
            p.vT[(((size_t)(b * H_ + h)) * HS_ + e) * T_ + t] = bv;
        } else if constexpr (MODE == 1) {       // proj: +bias +residual -> fp32
          const size_t o = (size_t)gm * p.N + gn;
          p.outF[o] = v + p.bias[gn] + p.resid[o];
        } else if constexpr (MODE == 2) {       // FFN1: relu(+bias) -> bf16
          p.outB[(size_t)gm * p.N + gn] = (bf16)fmaxf(v + p.bias[gn], 0.f);
        } else {                                // FFN2: gated accumulate
          const float g = p.gate[(size_t)gm * E_ + p.expert];
          const size_t o = (size_t)gm * p.N + gn;
          p.outF[o] += g * (v + p.bias[gn]);
        }
      }
    }
  }
}

// ---------------------------------------------------------------------------
// Flash attention: one block = 128 q rows of one (b,h); 8 waves x 16 rows.
// S = q k^T * D^-0.5 (reference scales by n_embd^-0.5 = 1/32), online softmax,
// O += P V with vT pre-transposed so B-fragments are natural rows.
// ---------------------------------------------------------------------------
__global__ void __launch_bounds__(256) k_attn(const bf16* __restrict__ q,
                                              const bf16* __restrict__ k,
                                              const bf16* __restrict__ vT,
                                              bf16* __restrict__ attnb) {
  __shared__ __align__(16) bf16 Qs[128 * 64];
  __shared__ __align__(16) bf16 Ks[32 * 64];
  __shared__ __align__(16) bf16 Vs[64 * 32];
  __shared__ __align__(16) bf16 Ps[8 * 16 * 32];

  const int bh = blockIdx.y;
  const int b = bh >> 4, h = bh & 15;
  const int q0 = blockIdx.x * 128;
  const int tid = threadIdx.x, wid = tid >> 5, lane = tid & 31;
  const int l16 = lane & 15, hf = lane >> 4;

  const bf16* qg = q  + ((size_t)bh * T_ + q0) * HS_;
  const bf16* kg = k  + (size_t)bh * T_ * HS_;
  const bf16* vg = vT + (size_t)bh * HS_ * T_;

  {  // stage Q block (128x64) via async copies: 64B per thread
    const int row = tid >> 1, c = (tid & 1) * 32;
    async_b128_x4(lds_addr32(Qs + row * 64 + c), qg + row * 64 + c);
  }
  const int krow = tid >> 3, kc = (tid & 7) * 8;    // Ks: 32x64
  const int vrow = tid >> 2, vc = (tid & 3) * 8;    // Vs: 64x32 (vT rows)
  const uint32_t ldsK = lds_addr32(Ks + krow * 64 + kc);
  const uint32_t ldsV = lds_addr32(Vs + vrow * 32 + vc);

  v8f o[4];
#pragma unroll
  for (int j = 0; j < 4; ++j)
#pragma unroll
    for (int r = 0; r < 8; ++r) o[j][r] = 0.f;
  float mrow[8], lrow[8];
#pragma unroll
  for (int r = 0; r < 8; ++r) { mrow[r] = -3e38f; lrow[r] = 0.f; }

  bf16* pp = Ps + wid * (16 * 32);

  for (int kv = 0; kv < T_; kv += 32) {
    __syncthreads();                    // prev compute done reading Ks/Vs
    async_b128(ldsK, kg + (size_t)(kv + krow) * 64 + kc);
    async_b128(ldsV, vg + (size_t)vrow * T_ + kv + vc);
    wait_async<0>();
    __syncthreads();

    v8f s2[2];
#pragma unroll
    for (int sub = 0; sub < 2; ++sub) {
      v8f s;
#pragma unroll
      for (int r = 0; r < 8; ++r) s[r] = 0.f;
#pragma unroll
      for (int kcc = 0; kcc < 64; kcc += 32) {
        v16bf a  = frag_a(Qs + kcc, wid * 16, 64, lane);
        v16bf bb = frag_b(Ks + kcc, sub * 16, 64, lane);
        s = wmma_bf16(a, bb, s);
      }
#pragma unroll
      for (int r = 0; r < 8; ++r) s[r] *= 0.03125f;   // D^-0.5
      s2[sub] = s;
    }

    float alpha[8];
#pragma unroll
    for (int r = 0; r < 8; ++r) {
      float mx = fmaxf(s2[0][r], s2[1][r]);
#pragma unroll
      for (int msk = 1; msk < 16; msk <<= 1)
        mx = fmaxf(mx, __shfl_xor(mx, msk, 32));
      const float mnew = fmaxf(mrow[r], mx);
      alpha[r] = expf(mrow[r] - mnew);
      const float e0 = expf(s2[0][r] - mnew);
      const float e1 = expf(s2[1][r] - mnew);
      s2[0][r] = e0; s2[1][r] = e1;
      float sm = e0 + e1;
#pragma unroll
      for (int msk = 1; msk < 16; msk <<= 1)
        sm += __shfl_xor(sm, msk, 32);
      lrow[r] = lrow[r] * alpha[r] + sm;
      mrow[r] = mnew;
    }
#pragma unroll
    for (int j = 0; j < 4; ++j)
#pragma unroll
      for (int r = 0; r < 8; ++r) o[j][r] *= alpha[r];

    // Re-shape P from C-layout to A-layout through per-wave LDS staging.
#pragma unroll
    for (int r = 0; r < 8; ++r) {
      pp[(r + hf * 8) * 32 + l16]      = (bf16)s2[0][r];
      pp[(r + hf * 8) * 32 + 16 + l16] = (bf16)s2[1][r];
    }
    __syncthreads();
    v16bf a = frag_a(pp, 0, 32, lane);
#pragma unroll
    for (int j = 0; j < 4; ++j) {
      v16bf bb = frag_b(Vs, j * 16, 32, lane);
      o[j] = wmma_bf16(a, bb, o[j]);
    }
  }

#pragma unroll
  for (int r = 0; r < 8; ++r) {
    const float inv = 1.f / lrow[r];
    const int t = q0 + wid * 16 + r + hf * 8;
    const size_t rowo = ((size_t)(b * T_ + t)) * D_ + h * HS_;
#pragma unroll
    for (int j = 0; j < 4; ++j)
      attnb[rowo + j * 16 + l16] = (bf16)(o[j][r] * inv);
  }
}

// ---------------------------------------------------------------------------
// LayerNorm: one wave per token (D=1024 -> 32 elems/lane).
// ---------------------------------------------------------------------------
__global__ void __launch_bounds__(256) k_ln(const float* __restrict__ x,
                                            const float* __restrict__ gw,
                                            const float* __restrict__ bw,
                                            bf16* __restrict__ outB,
                                            float* __restrict__ outF) {
  const int token = blockIdx.x * 8 + (threadIdx.x >> 5);
  const int lane = threadIdx.x & 31;
  const float* row = x + (size_t)token * D_;
  float vals[32];
  float s = 0.f, ss = 0.f;
#pragma unroll
  for (int i = 0; i < 32; ++i) {
    const float v = row[lane + 32 * i];
    vals[i] = v; s += v; ss += v * v;
  }
#pragma unroll
  for (int msk = 1; msk < 32; msk <<= 1) {
    s += __shfl_xor(s, msk, 32);
    ss += __shfl_xor(ss, msk, 32);
  }
  const float mu = s * (1.f / D_);
  const float var = ss * (1.f / D_) - mu * mu;
  const float rstd = rsqrtf(var + 1e-5f);
#pragma unroll
  for (int i = 0; i < 32; ++i) {
    const int d = lane + 32 * i;
    const float v = (vals[i] - mu) * rstd * gw[d] + bw[d];
    outB[(size_t)token * D_ + d] = (bf16)v;
    if (outF) outF[(size_t)token * D_ + d] = v;
  }
}

// ---------------------------------------------------------------------------
// Router: noisy top-2 gating, one wave per token.
// ---------------------------------------------------------------------------
__global__ void __launch_bounds__(256) k_route(const float* __restrict__ xn2,
                                               const float* __restrict__ wr,
                                               const float* __restrict__ br,
                                               const float* __restrict__ wn,
                                               const float* __restrict__ bn,
                                               const float* __restrict__ noise,
                                               float* __restrict__ gating) {
  const int token = blockIdx.x * 8 + (threadIdx.x >> 5);
  const int lane = threadIdx.x & 31;
  const float* row = xn2 + (size_t)token * D_;
  float aR[E_], aN[E_];
#pragma unroll
  for (int e = 0; e < E_; ++e) { aR[e] = 0.f; aN[e] = 0.f; }
  for (int i = 0; i < 32; ++i) {
    const int d = lane + 32 * i;
    const float xv = row[d];
    const float* wrd = wr + (size_t)d * E_;
    const float* wnd = wn + (size_t)d * E_;
#pragma unroll
    for (int e = 0; e < E_; ++e) { aR[e] += xv * wrd[e]; aN[e] += xv * wnd[e]; }
  }
#pragma unroll
  for (int e = 0; e < E_; ++e)
#pragma unroll
    for (int msk = 1; msk < 32; msk <<= 1) {
      aR[e] += __shfl_xor(aR[e], msk, 32);
      aN[e] += __shfl_xor(aN[e], msk, 32);
    }
  if (lane == 0) {
    float noisy[E_];
#pragma unroll
    for (int e = 0; e < E_; ++e) {
      const float nl = aN[e] + bn[e];
      const float sp = (nl > 20.f) ? nl : log1pf(expf(nl));
      noisy[e] = aR[e] + br[e] + noise[(size_t)token * E_ + e] * sp;
    }
    int i0 = 0;
#pragma unroll
    for (int e = 1; e < E_; ++e) if (noisy[e] > noisy[i0]) i0 = e;  // first idx ties
    int i1 = (i0 == 0) ? 1 : 0;
#pragma unroll
    for (int e = 0; e < E_; ++e)
      if (e != i0 && noisy[e] > noisy[i1]) i1 = e;
    const float p0 = 1.f / (1.f + expf(noisy[i1] - noisy[i0]));
    const float p1 = 1.f - p0;
#pragma unroll
    for (int e = 0; e < E_; ++e)
      gating[(size_t)token * E_ + e] = (e == i0) ? p0 : ((e == i1) ? p1 : 0.f);
  }
}

// ---------------------------------------------------------------------------
// Weight packing: fp32 -> bf16, transposed to [N][K] for B-fragment loads.
// ---------------------------------------------------------------------------
__global__ void __launch_bounds__(256) k_pack_qkv(const float* __restrict__ wq,
                                                  const float* __restrict__ wk,
                                                  const float* __restrict__ wv,
                                                  bf16* __restrict__ dst) {
  const int idx = blockIdx.x * 256 + threadIdx.x;   // [3072][1024]
  const int n = idx >> 10, d = idx & 1023;
  const int sec = n >> 10, ns = n & 1023, h = ns >> 6, e = ns & 63;
  const float* w = (sec == 0) ? wq : ((sec == 1) ? wk : wv);
  dst[idx] = (bf16)w[((size_t)h * D_ + d) * HS_ + e];
}

__global__ void k_packT(const float* __restrict__ src, bf16* __restrict__ dst,
                        int K, int N) {
  __shared__ float tile[32][33];
  const float* s = src + (size_t)blockIdx.z * K * N;
  bf16* d = dst + (size_t)blockIdx.z * K * N;
  const int n0 = blockIdx.x * 32, k0 = blockIdx.y * 32;
  for (int i = threadIdx.y; i < 32; i += 8)
    tile[i][threadIdx.x] = s[(size_t)(k0 + i) * N + n0 + threadIdx.x];
  __syncthreads();
  for (int i = threadIdx.y; i < 32; i += 8)
    d[(size_t)(n0 + i) * K + k0 + threadIdx.x] = (bf16)tile[threadIdx.x][i];
}

// ---------------------------------------------------------------------------
extern "C" void kernel_launch(void* const* d_in, const int* in_sizes, int n_in,
                              void* d_out, int out_size, void* d_ws, size_t ws_size,
                              hipStream_t stream) {
  const float* x       = (const float*)d_in[0];
  const float* noise   = (const float*)d_in[1];
  const float* ln1_g   = (const float*)d_in[2];
  const float* ln1_b   = (const float*)d_in[3];
  const float* wq      = (const float*)d_in[4];
  const float* wk      = (const float*)d_in[5];
  const float* wv      = (const float*)d_in[6];
  const float* w_proj  = (const float*)d_in[7];
  const float* b_proj  = (const float*)d_in[8];
  const float* ln2_g   = (const float*)d_in[9];
  const float* ln2_b   = (const float*)d_in[10];
  const float* w_route = (const float*)d_in[11];
  const float* b_route = (const float*)d_in[12];
  const float* w_noise = (const float*)d_in[13];
  const float* b_noise = (const float*)d_in[14];
  const float* w1      = (const float*)d_in[15];
  const float* b1      = (const float*)d_in[16];
  const float* w2      = (const float*)d_in[17];
  const float* b2      = (const float*)d_in[18];
  float* out = (float*)d_out;

  char* wsp = (char*)d_ws;
  auto alloc = [&](size_t bytes) -> void* {
    void* p = (void*)wsp;
    wsp += (bytes + 255) & ~(size_t)255;
    return p;
  };
  bf16*  xn     = (bf16*)alloc((size_t)BT_ * D_ * 2);
  bf16*  xn2b   = (bf16*)alloc((size_t)BT_ * D_ * 2);
  bf16*  qb     = (bf16*)alloc((size_t)BT_ * D_ * 2);
  bf16*  kb     = (bf16*)alloc((size_t)BT_ * D_ * 2);
  bf16*  vTb    = (bf16*)alloc((size_t)BT_ * D_ * 2);
  bf16*  attnb  = (bf16*)alloc((size_t)BT_ * D_ * 2);
  float* x2     = (float*)alloc((size_t)BT_ * D_ * 4);
  float* gating = (float*)alloc((size_t)BT_ * E_ * 4);
  bf16*  wqkvT  = (bf16*)alloc((size_t)3 * D_ * D_ * 2);
  bf16*  wprojT = (bf16*)alloc((size_t)D_ * D_ * 2);
  bf16*  w1T    = (bf16*)alloc((size_t)E_ * D_ * DFF_ * 2);
  bf16*  w2T    = (bf16*)alloc((size_t)E_ * D_ * DFF_ * 2);
  bf16*  hbuf   = (bf16*)alloc((size_t)BT_ * DFF_ * 2);

  // ---- pack weights to bf16 [N][K] ----
  k_pack_qkv<<<(3 * D_ * D_) / 256, 256, 0, stream>>>(wq, wk, wv, wqkvT);
  k_packT<<<dim3(D_ / 32, D_ / 32, 1), dim3(32, 8), 0, stream>>>(w_proj, wprojT, D_, D_);
  k_packT<<<dim3(DFF_ / 32, D_ / 32, E_), dim3(32, 8), 0, stream>>>(w1, w1T, D_, DFF_);
  k_packT<<<dim3(D_ / 32, DFF_ / 32, E_), dim3(32, 8), 0, stream>>>(w2, w2T, DFF_, D_);

  // ---- LN1 ----
  k_ln<<<BT_ / 8, 256, 0, stream>>>(x, ln1_g, ln1_b, xn, (float*)nullptr);

  // ---- fused QKV GEMM ----
  GemmP pq{};
  pq.A = xn; pq.Bt = wqkvT; pq.M = BT_; pq.N = 3 * D_; pq.K = D_;
  pq.q = qb; pq.k = kb; pq.vT = vTb;
  k_gemm<0><<<dim3(3 * D_ / 128, BT_ / 128), 256, 0, stream>>>(pq);

  // ---- flash attention ----
  k_attn<<<dim3(T_ / 128, B_ * H_), 256, 0, stream>>>(qb, kb, vTb, attnb);

  // ---- output projection + bias + residual ----
  GemmP pp{};
  pp.A = attnb; pp.Bt = wprojT; pp.M = BT_; pp.N = D_; pp.K = D_;
  pp.bias = b_proj; pp.resid = x; pp.outF = x2;
  k_gemm<1><<<dim3(D_ / 128, BT_ / 128), 256, 0, stream>>>(pp);

  // ---- LN2: bf16 for GEMMs + fp32 xn2 into d_out (output = xn2 + moe) ----
  k_ln<<<BT_ / 8, 256, 0, stream>>>(x2, ln2_g, ln2_b, xn2b, out);

  // ---- noisy top-2 routing (reads fp32 xn2 from d_out, before FFN writes) ----
  k_route<<<BT_ / 8, 256, 0, stream>>>(out, w_route, b_route, w_noise, b_noise,
                                       noise, gating);

  // ---- dense all-expert FFN (gating==0 for unselected -> exact) ----
  for (int e = 0; e < E_; ++e) {
    GemmP p1{};
    p1.A = xn2b; p1.Bt = w1T + (size_t)e * D_ * DFF_;
    p1.M = BT_; p1.N = DFF_; p1.K = D_;
    p1.bias = b1 + (size_t)e * DFF_; p1.outB = hbuf;
    k_gemm<2><<<dim3(DFF_ / 128, BT_ / 128), 256, 0, stream>>>(p1);

    GemmP p2{};
    p2.A = hbuf; p2.Bt = w2T + (size_t)e * D_ * DFF_;
    p2.M = BT_; p2.N = D_; p2.K = DFF_;
    p2.bias = b2 + (size_t)e * D_; p2.gate = gating; p2.expert = e; p2.outF = out;
    k_gemm<3><<<dim3(D_ / 128, BT_ / 128), 256, 0, stream>>>(p2);
  }

  (void)in_sizes; (void)n_in; (void)out_size; (void)ws_size;
}
